// LambdaLayer_22746146799768
// MI455X (gfx1250) — matrix operand order repaired and checked by reading, not compiled, not run on hardware
//
#include <hip/hip_runtime.h>

typedef __attribute__((ext_vector_type(16))) _Float16 v16h;
typedef __attribute__((ext_vector_type(8)))  _Float16 v8h;
typedef __attribute__((ext_vector_type(8)))  float    v8f;

// Problem constants (fixed by the reference)
constexpr int kBATCH = 16;
constexpr int kC     = 256;
constexpr int kN     = 1024;   // 32x32
constexpr int kNT    = 4;      // query positions per workgroup in main kernel
constexpr int kMC    = 128;    // m-chunk staged through LDS per iteration

// ---------------------------------------------------------------------------
// Kernel A: fused 1x1-conv projections + affine + K-softmax.
// Grid: 16 b * 9 channel-blocks(16ch) * 4 n-tiles(256) = 576 WGs of 256 thr.
// W reads are thread-uniform -> scalar loads; x reads coalesced b32.
// Outputs (all f16 except sigma):
//   Qh [n][bh=4b+h][k]  (A-tiles: 16B-contiguous per half-K)
//   Vht[b][v][m]        (B-tiles: 32B-contiguous per lane)
//   sigma[b][m][k] f32
// ---------------------------------------------------------------------------
__global__ __launch_bounds__(256) void proj_kernel(
    const float* __restrict__ x,  const float* __restrict__ Wq,
    const float* __restrict__ gq, const float* __restrict__ bq,
    const float* __restrict__ Wk, const float* __restrict__ Wv,
    const float* __restrict__ gv, const float* __restrict__ bv,
    _Float16* __restrict__ Qh, _Float16* __restrict__ Vht,
    float* __restrict__ sigma)
{
    const int wg = blockIdx.x;
    const int b  = wg / 36;
    const int ob = (wg / 4) % 9;       // 0..3 Q, 4 K, 5..8 V
    const int nt = wg % 4;
    const int n  = nt * 256 + threadIdx.x;

    const float* Wp;
    if (ob < 4)       Wp = Wq + ob * 16 * kC;
    else if (ob == 4) Wp = Wk;
    else              Wp = Wv + (ob - 5) * 16 * kC;

    float acc[16];
#pragma unroll
    for (int r = 0; r < 16; ++r) acc[r] = 0.f;

    const float* xp = x + (size_t)b * kC * kN + n;
    for (int c = 0; c < kC; ++c) {
        const float xv = xp[c * kN];
#pragma unroll
        for (int r = 0; r < 16; ++r) acc[r] = fmaf(Wp[r * kC + c], xv, acc[r]);
    }

    if (ob < 4) {                       // Q + affine
#pragma unroll
        for (int r = 0; r < 16; ++r) {
            const int o = ob * 16 + r;  // h = ob, k = r
            Qh[((n * 64) + b * 4 + ob) * 16 + r] = (_Float16)(gq[o] * acc[r] + bq[o]);
        }
    } else if (ob == 4) {               // K -> softmax over 16 k's
        float mx = acc[0];
#pragma unroll
        for (int r = 1; r < 16; ++r) mx = fmaxf(mx, acc[r]);
        float e[16], s = 0.f;
#pragma unroll
        for (int r = 0; r < 16; ++r) { e[r] = __expf(acc[r] - mx); s += e[r]; }
        const float inv = 1.f / s;
#pragma unroll
        for (int r = 0; r < 16; ++r) sigma[(b * kN + n) * 16 + r] = e[r] * inv;
    } else {                            // V + affine, stored v-major
#pragma unroll
        for (int r = 0; r < 16; ++r) {
            const int v = (ob - 5) * 16 + r;
            Vht[(b * 64 + v) * kN + n] = (_Float16)(gv[v] * acc[r] + bv[v]);
        }
    }
}

// ---------------------------------------------------------------------------
// Kernel B: rel_pos_emb (63*63*16 f32) -> f16 table (rows = 32B, L2 resident)
// ---------------------------------------------------------------------------
__global__ void emb_cvt(const float* __restrict__ E, _Float16* __restrict__ Eh, int ntot)
{
    const int i = blockIdx.x * 256 + threadIdx.x;
    if (i < ntot) Eh[i] = (_Float16)E[i];
}

// ---------------------------------------------------------------------------
// Kernel C: lam_c[b,k,v] = sum_m sigma[b,m,k] * V[b,m,v]; stored v-major f16
// (16.8M MACs total -> plain VALU is fine)
// ---------------------------------------------------------------------------
__global__ __launch_bounds__(256) void lamc_kernel(
    const float* __restrict__ sigma, const _Float16* __restrict__ Vht,
    _Float16* __restrict__ Lh)
{
    const int b = blockIdx.x;
#pragma unroll
    for (int p = 0; p < 4; ++p) {
        const int vk = threadIdx.x + p * 256;
        const int v = vk >> 4, k = vk & 15;
        float acc = 0.f;
        const float*    sp = sigma + (size_t)b * kN * 16 + k;
        const _Float16* vp = Vht + (b * 64 + v) * kN;
        for (int m = 0; m < kN; ++m) acc = fmaf(sp[m * 16], (float)vp[m], acc);
        Lh[(b * 64 + v) * 16 + k] = (_Float16)acc;
    }
}

// ---------------------------------------------------------------------------
// Kernel D (main): per WG handle 4 query positions n, all 16 batches.
//  Phase 1 (scores): wave (sn,st) does S[bh-tile st, n_base+sn, m] via
//     v_wmma_f32_16x16x32_f16 with A=Q_n (K padded to 32), B=E_n^T gathered
//     from the f16 table; results restaged into LDS as S[b][(nl,h)][m] f16.
//  Phase 2 (PV): wave b accumulates C += S_b(16x32) @ V_b(32x16), full K=32.
//  Content term folded in as one extra K-padded WMMA per v-tile (B=lam_c).
// ---------------------------------------------------------------------------
__global__ __launch_bounds__(512) void lambda_main(
    const _Float16* __restrict__ Qh, const _Float16* __restrict__ Vht,
    const _Float16* __restrict__ Eh, const _Float16* __restrict__ Lh,
    float* __restrict__ out)
{
    __shared__ __align__(32) _Float16 S[16][16][kMC];   // [b][nl*4+h][m] 64KB

    const int tid  = threadIdx.x;
    const int wave = tid >> 5;
    const int lane = tid & 31;
    const int row  = lane & 15;
    const int hi   = lane >> 4;            // half-wave select
    const int n_base = blockIdx.x * kNT;

    // ---- scores role: wave -> (n_local, bh row-tile) ----
    const int sn = wave >> 2;
    const int st = wave & 3;
    const int n  = n_base + sn;
    const int xn = n >> 5, yn = n & 31;

    // A = Q_n rows bh=st*16+row, real K=0..15 in lo lanes (hi lanes = K 16..31 zero-pad)
    v16h As = {};
    {
        if (hi == 0) As = *(const v16h*)(Qh + ((n * 64) + st * 16 + row) * 16);
    }

    // ---- PV role: wave -> batch b, 4 v-tile fp32 accumulators ----
    const int b = wave;
    v8f acc[4];
    {
        v8f z = {};
#pragma unroll
        for (int t = 0; t < 4; ++t) acc[t] = z;
    }

    for (int ch = 0; ch < kN / kMC; ++ch) {
        const int mbase = ch * kMC;

        // -------- phase 1: scores for this m-chunk --------
#pragma unroll
        for (int mt = 0; mt < kMC / 16; ++mt) {
            v16h Bs = {};                      // hi lanes = K 16..31 zero-pad
            if (hi == 0) {
                const int m  = mbase + mt * 16 + row;              // column
                const int idx = ((m >> 5) - xn + 31) * 63 + ((m & 31) - yn + 31);
                Bs = *(const v16h*)(Eh + idx * 16);                // 32B row
            }
            v8f c = {};
            c = __builtin_amdgcn_wmma_f32_16x16x32_f16(
                    false, As, false, Bs, (short)0, c, false, false);
            // restage: C row M -> (b = st*4 + M/4, lds row = sn*4 + M%4)
#pragma unroll
            for (int r = 0; r < 8; ++r) {
                const int M = r + hi * 8;
                S[st * 4 + (M >> 2)][sn * 4 + (M & 3)][mt * 16 + row] = (_Float16)c[r];
            }
        }
        __syncthreads();

        // -------- phase 2: out += S_b @ V_b (full K=32 per WMMA) --------
#pragma unroll
        for (int kt = 0; kt < kMC / 32; ++kt) {
            v16h Ap;
            {
                Ap = *(const v16h*)&S[b][row][kt * 32 + hi * 16];
            }
#pragma unroll
            for (int vt = 0; vt < 4; ++vt) {
                const v16h Bp = *(const v16h*)(
                    Vht + ((b * 64) + vt * 16 + row) * kN + mbase + kt * 32 + hi * 16);
                acc[vt] = __builtin_amdgcn_wmma_f32_16x16x32_f16(
                              false, Ap, false, Bp, (short)0, acc[vt], false, false);
            }
        }
        __syncthreads();
    }

    // -------- content term: C += Q_n(16x16 pad32) @ lam_c_b --------
    {
        v16h Ac = {};
        if (hi == 0) Ac = *(const v16h*)(
            Qh + ((n_base + (row >> 2)) * 64 + b * 4 + (row & 3)) * 16);
#pragma unroll
        for (int vt = 0; vt < 4; ++vt) {
            v16h Bc = {};
            if (hi == 0) Bc = *(const v16h*)(Lh + ((b * 64) + vt * 16 + row) * 16);
            acc[vt] = __builtin_amdgcn_wmma_f32_16x16x32_f16(
                          false, Ac, false, Bc, (short)0, acc[vt], false, false);
        }
    }

    // -------- epilogue: C rows M=(nl*4+h), cols v=vt*16+lane%16 --------
#pragma unroll
    for (int vt = 0; vt < 4; ++vt) {
#pragma unroll
        for (int r = 0; r < 8; ++r) {
            const int M  = r + hi * 8;
            const int nl = M >> 2, h = M & 3;
            out[((size_t)b * kN + n_base + nl) * 256 + h * 64 + vt * 16 + row] = acc[vt][r];
        }
    }
}

// ---------------------------------------------------------------------------
extern "C" void kernel_launch(void* const* d_in, const int* in_sizes, int n_in,
                              void* d_out, int out_size, void* d_ws, size_t ws_size,
                              hipStream_t stream)
{
    const float* x   = (const float*)d_in[0];
    const float* Wq  = (const float*)d_in[1];
    const float* gq  = (const float*)d_in[2];
    const float* bq  = (const float*)d_in[3];
    const float* Wk  = (const float*)d_in[4];
    const float* Wv  = (const float*)d_in[5];
    const float* gv  = (const float*)d_in[6];
    const float* bv  = (const float*)d_in[7];
    const float* rpe = (const float*)d_in[8];
    float* out = (float*)d_out;

    // Workspace layout (~5.2 MB total)
    char* ws = (char*)d_ws;
    _Float16* Qh  = (_Float16*)(ws);                         // 1024*64*16*2 = 2 MB
    _Float16* Vht = (_Float16*)(ws + (2u << 20));            // 16*64*1024*2 = 2 MB
    float*    sig = (float*)   (ws + (4u << 20));            // 16*1024*16*4 = 1 MB
    _Float16* Eh  = (_Float16*)(ws + (5u << 20));            // 63*63*16*2 = 124 KB
    _Float16* Lh  = (_Float16*)(ws + (5u << 20) + 131072);   // 16*64*16*2 = 32 KB

    proj_kernel<<<576, 256, 0, stream>>>(x, Wq, gq, bq, Wk, Wv, gv, bv, Qh, Vht, sig);
    emb_cvt<<<(63 * 63 * 16 + 255) / 256, 256, 0, stream>>>(rpe, Eh, 63 * 63 * 16);
    lamc_kernel<<<kBATCH, 256, 0, stream>>>(sig, Vht, Lh);
    lambda_main<<<kN / kNT, 512, 0, stream>>>(Qh, Vht, Eh, Lh, out);
}
